// ExpertParallelFrontBlock_71571335020919
// MI455X (gfx1250) — matrix operand order repaired and checked by reading, not compiled, not run on hardware
//
#include <hip/hip_runtime.h>
#include <hip/hip_bf16.h>
#include <stdint.h>

typedef __attribute__((ext_vector_type(16))) __bf16 v16bf;
typedef __attribute__((ext_vector_type(8)))  __bf16 v8bf;
typedef __attribute__((ext_vector_type(4)))  __bf16 v4bf;
typedef __attribute__((ext_vector_type(8)))  float  v8f;
typedef int v4i_ __attribute__((vector_size(16)));

#define TOKENS   8192
#define DMODEL   1024
#define DFF      4096
#define NEXP     8
#define CAP      1280
#define NSLOTS   (NEXP * CAP)                   // 10240
#define CW_ELEMS ((long long)TOKENS * NSLOTS)   // 83,886,080
#define OUT_ELEMS ((long long)NEXP * CAP * DFF) // 41,943,040

// ---------------------------------------------------------------------------
// gfx1250 async global->LDS copy (ASYNCcnt path). Builtin if present,
// inline-asm fallback otherwise.
// ---------------------------------------------------------------------------
__device__ __forceinline__ void async_copy_b128(const void* g, void* lds) {
#if defined(__has_builtin) && __has_builtin(__builtin_amdgcn_global_load_async_to_lds_b128)
  __builtin_amdgcn_global_load_async_to_lds_b128(
      (__attribute__((address_space(1))) v4i_*)g,
      (__attribute__((address_space(3))) v4i_*)lds, 0, 0);
#else
  asm volatile("global_load_async_to_lds_b128 %0, %1, off"
               :: "v"((unsigned)(uintptr_t)lds),
                  "v"((unsigned long long)(uintptr_t)g)
               : "memory");
#endif
}

__device__ __forceinline__ void wait_async0() {
#if defined(__has_builtin) && __has_builtin(__builtin_amdgcn_s_wait_asynccnt)
  __builtin_amdgcn_s_wait_asynccnt(0);
#else
  asm volatile("s_wait_asynccnt 0x0" ::: "memory");
#endif
}

__device__ __forceinline__ __bf16 hi_bf(float a) { return (__bf16)a; }
__device__ __forceinline__ __bf16 lo_bf(float a, __bf16 h) {
  return (__bf16)(a - (float)h);
}

// assemble a v16bf fragment from two contiguous 16B LDS runs
__device__ __forceinline__ v16bf frag16(const __bf16* p0, const __bf16* p1) {
  v8bf a = *(const v8bf*)p0;
  v8bf b = *(const v8bf*)p1;
  return __builtin_shufflevector(a, b, 0, 1, 2, 3, 4, 5, 6, 7,
                                 8, 9, 10, 11, 12, 13, 14, 15);
}

// ---------------------------------------------------------------------------
// Kernel 0: zero combine_weights region, init slot table to -1
// ---------------------------------------------------------------------------
__global__ void init_kernel(float* __restrict__ cw, long long n4,
                            int* __restrict__ slots, int nslots) {
  long long i = (long long)blockIdx.x * blockDim.x + threadIdx.x;
  long long stride = (long long)gridDim.x * blockDim.x;
  float4 z; z.x = 0.f; z.y = 0.f; z.z = 0.f; z.w = 0.f;
  for (long long j = i; j < n4; j += stride) ((float4*)cw)[j] = z;
  if (i < nslots) slots[i] = -1;
}

// ---------------------------------------------------------------------------
// Kernel 1: fp32 gate. One wave per token; gate weights async-staged in LDS.
// ---------------------------------------------------------------------------
__global__ __launch_bounds__(256) void gate_kernel(
    const float* __restrict__ x, const float* __restrict__ gw,
    int* __restrict__ top1, float* __restrict__ gate_w,
    float* __restrict__ wsme) {
  __shared__ __align__(16) float sgw[NEXP * DMODEL];   // 32 KB
  __shared__ float wprob[8][NEXP];
  int tid = threadIdx.x, lane = tid & 31, wid = tid >> 5;
  for (int i = tid; i < NEXP * DMODEL / 4; i += 256)
    async_copy_b128(gw + i * 4, &sgw[i * 4]);
  wait_async0();
  __syncthreads();

  int token = blockIdx.x * 8 + wid;
  const float* xr = x + (size_t)token * DMODEL;
  float acc[NEXP];
#pragma unroll
  for (int e = 0; e < NEXP; ++e) acc[e] = 0.f;
  for (int d = lane; d < DMODEL; d += 32) {
    float xv = xr[d];
#pragma unroll
    for (int e = 0; e < NEXP; ++e) acc[e] += xv * sgw[e * DMODEL + d];
  }
#pragma unroll
  for (int e = 0; e < NEXP; ++e)
    for (int off = 16; off; off >>= 1) acc[e] += __shfl_xor(acc[e], off, 32);

  float mx = acc[0]; int ai = 0;
#pragma unroll
  for (int e = 1; e < NEXP; ++e) if (acc[e] > mx) { mx = acc[e]; ai = e; }
  float pr[NEXP]; float den = 0.f;
#pragma unroll
  for (int e = 0; e < NEXP; ++e) { pr[e] = __expf(acc[e] - mx); den += pr[e]; }
  float inv = 1.f / den;
  if (lane == 0) {
    top1[token] = ai;
    gate_w[token] = pr[ai] * inv;
#pragma unroll
    for (int e = 0; e < NEXP; ++e) wprob[wid][e] = pr[e] * inv;
  }
  __syncthreads();
  if (tid < NEXP) {                 // fixed-order per-block sum: deterministic
    float s = 0.f;
    for (int w = 0; w < 8; ++w) s += wprob[w][tid];
    wsme[blockIdx.x * NEXP + tid] = s;
  }
}

// ---------------------------------------------------------------------------
// Kernel 2: deterministic cumsum capacity ranking (single workgroup),
// ballot scan per chunk; emits slot table, dest indices, l_aux.
// ---------------------------------------------------------------------------
__global__ __launch_bounds__(256) void rank_kernel(
    const int* __restrict__ top1, const float* __restrict__ wsme,
    int* __restrict__ slot_to_token, int* __restrict__ dest_cw,
    float* __restrict__ l_aux_out) {
  __shared__ float me[NEXP];
  __shared__ int base[NEXP];
  __shared__ int wcnt[8][NEXP];
  int tid = threadIdx.x, lane = tid & 31, wid = tid >> 5;

  {
    float s = 0.f;
    for (int b = lane; b < TOKENS / 8; b += 32) s += wsme[b * NEXP + wid];
    for (int off = 16; off; off >>= 1) s += __shfl_xor(s, off, 32);
    if (lane == 0) me[wid] = s / (float)TOKENS;
  }
  if (tid < NEXP) base[tid] = 0;
  __syncthreads();

  for (int c0 = 0; c0 < TOKENS; c0 += 256) {
    int t = c0 + tid;
    int e = top1[t];
    unsigned bm[NEXP];
#pragma unroll
    for (int v = 0; v < NEXP; ++v) bm[v] = (unsigned)__ballot(e == v);
    int laneoff = __popc(bm[e] & ((1u << lane) - 1u));
    if (lane == 0) {
#pragma unroll
      for (int v = 0; v < NEXP; ++v) wcnt[wid][v] = __popc(bm[v]);
    }
    __syncthreads();
    int waveoff = 0;
    for (int w = 0; w < wid; ++w) waveoff += wcnt[w][e];
    int rank = base[e] + waveoff + laneoff;
    if (rank < CAP) {
      int slot = e * CAP + rank;
      slot_to_token[slot] = t;
      dest_cw[t] = slot;
    } else {
      dest_cw[t] = -1;
    }
    __syncthreads();
    if (tid < NEXP) {
      int tot = 0;
      for (int w = 0; w < 8; ++w) tot += wcnt[w][tid];
      base[tid] += tot;
    }
    __syncthreads();
  }
  if (tid == 0) {
    float l = 0.f;
    for (int e = 0; e < NEXP; ++e)
      l += me[e] * ((float)base[e] / (float)TOKENS);
    *l_aux_out = (float)NEXP * l;
  }
}

// ---------------------------------------------------------------------------
// Kernel 3: scatter nonzero combine weights
// ---------------------------------------------------------------------------
__global__ void scatter_cw(const int* __restrict__ dest_cw,
                           const float* __restrict__ gate_w,
                           float* __restrict__ cw) {
  int t = blockIdx.x * blockDim.x + threadIdx.x;
  if (t < TOKENS) {
    int d = dest_cw[t];
    if (d >= 0) cw[(size_t)t * NSLOTS + d] = gate_w[t];
  }
}

// ---------------------------------------------------------------------------
// Kernel 4: expert GEMM, 128x128x32 tiles.
//   * fp32 staging tiles double-buffered, filled by async global->LDS b128
//     (A tile is a per-row token gather; empty slots -> zero rows)
//   * one conversion pass per element: fp32 -> (hi,lo) bf16 LDS tiles,
//     B stored transposed so fragments are contiguous b128 pairs
//   * 4x2 fp32 16x16 accumulators per wave, 3 chained
//     v_wmma_f32_16x16x32_bf16 per accumulator per K-step (split product)
// ---------------------------------------------------------------------------
#define TM 128
#define TN 128
#define TK 32
#define FR 40   // bf16 row stride: 80B rows, 16B aligned, conflict-free

__device__ __forceinline__ v8f wmma_bf16(v16bf a, v16bf b, v8f c) {
  return __builtin_amdgcn_wmma_f32_16x16x32_bf16(
      false, a, false, b, (short)0, c, false, false);
}

__global__ __launch_bounds__(256) void moe_gemm(
    const float* __restrict__ x,
    const float* __restrict__ W,      // [E][DMODEL][DFF]
    const float* __restrict__ bias,   // [E][DFF]
    const int* __restrict__ slot_to_token,
    float* __restrict__ out) {
  const int nTilesM = CAP / TM;       // 10
  const int nTilesN = DFF / TN;       // 32
  int bid = blockIdx.x;
  int e  = bid / (nTilesM * nTilesN);
  int r  = bid % (nTilesM * nTilesN);
  int tm = r / nTilesN, tn = r % nTilesN;

  __shared__ __align__(16) float  Afp[2][TM][TK];   // 32 KB (double buffer)
  __shared__ __align__(16) float  Bfp[2][TK][TN];   // 32 KB (double buffer)
  __shared__ __align__(16) __bf16 Ah[TM][FR], Al[TM][FR];     // 20 KB
  __shared__ __align__(16) __bf16 BhT[TN][FR], BlT[TN][FR];   // 20 KB
  __shared__ int tokRow[TM];

  int tid = threadIdx.x, lane = tid & 31, wid = tid >> 5;
  int wm = wid >> 2, wn = wid & 3;    // 2 x 4 wave grid
  int half = lane >> 4, lr = lane & 15;

  v8f acc[4][2];
#pragma unroll
  for (int i = 0; i < 4; ++i)
#pragma unroll
    for (int j = 0; j < 2; ++j)
#pragma unroll
      for (int v = 0; v < 8; ++v) acc[i][j][v] = 0.f;

  const float* We = W + (size_t)e * DMODEL * DFF;
  if (tid < TM) tokRow[tid] = slot_to_token[e * CAP + tm * TM + tid];
  __syncthreads();

  float4 f4z; f4z.x = 0.f; f4z.y = 0.f; f4z.z = 0.f; f4z.w = 0.f;

  auto issue = [&](int k0, int buf) {
#pragma unroll
    for (int t = 0; t < 4; ++t) {
      int idx = tid + t * 256;
      {  // A tile: gathered rows, 8 float4 per 32-wide row
        int row = idx >> 3, c4 = idx & 7;
        int tok = tokRow[row];
        if (tok >= 0)
          async_copy_b128(x + (size_t)tok * DMODEL + k0 + c4 * 4,
                          &Afp[buf][row][c4 * 4]);
        else
          *(float4*)&Afp[buf][row][c4 * 4] = f4z;
      }
      {  // B tile: 32 float4 per 128-wide row
        int row = idx >> 5, c4 = idx & 31;
        async_copy_b128(We + (size_t)(k0 + row) * DFF + tn * TN + c4 * 4,
                        &Bfp[buf][row][c4 * 4]);
      }
    }
  };

  issue(0, 0);
  int buf = 0;
  for (int k0 = 0; k0 < DMODEL; k0 += TK) {
    wait_async0();
    __syncthreads();                       // staged fp32 tiles visible

    // one fp32->(hi,lo) bf16 conversion per element
#pragma unroll
    for (int t = 0; t < 4; ++t) {
      int idx = tid + t * 256;
      int row = idx >> 3, c4 = idx & 7;
      float4 v = *(const float4*)&Afp[buf][row][c4 * 4];
      v4bf h, l;
      __bf16 hx = hi_bf(v.x); h[0] = hx; l[0] = lo_bf(v.x, hx);
      __bf16 hy = hi_bf(v.y); h[1] = hy; l[1] = lo_bf(v.y, hy);
      __bf16 hz = hi_bf(v.z); h[2] = hz; l[2] = lo_bf(v.z, hz);
      __bf16 hw = hi_bf(v.w); h[3] = hw; l[3] = lo_bf(v.w, hw);
      *(v4bf*)&Ah[row][c4 * 4] = h;
      *(v4bf*)&Al[row][c4 * 4] = l;
    }
#pragma unroll
    for (int t = 0; t < 4; ++t) {          // B: convert + transpose
      int idx = tid + t * 256;
      int row = idx >> 5, c4 = idx & 31;
      float4 v = *(const float4*)&Bfp[buf][row][c4 * 4];
      float vv[4] = {v.x, v.y, v.z, v.w};
#pragma unroll
      for (int c = 0; c < 4; ++c) {
        __bf16 h = hi_bf(vv[c]);
        BhT[c4 * 4 + c][row] = h;
        BlT[c4 * 4 + c][row] = lo_bf(vv[c], h);
      }
    }
    __syncthreads();                       // bf16 tiles ready

    if (k0 + TK < DMODEL) issue(k0 + TK, buf ^ 1);   // overlap next stage

    // fragments: contiguous b128 pairs per documented 16-bit layouts
    v16bf fAh[4], fAl[4], fBh[2], fBl[2];
#pragma unroll
    for (int i = 0; i < 4; ++i) {
      int m = wm * 64 + i * 16 + lr;
      fAh[i] = frag16(&Ah[m][half * 8], &Ah[m][16 + half * 8]);
      fAl[i] = frag16(&Al[m][half * 8], &Al[m][16 + half * 8]);
    }
#pragma unroll
    for (int j = 0; j < 2; ++j) {
      int n = wn * 32 + j * 16 + lr;
      fBh[j] = frag16(&BhT[n][half * 8], &BhT[n][16 + half * 8]);
      fBl[j] = frag16(&BlT[n][half * 8], &BlT[n][16 + half * 8]);
    }

#pragma unroll
    for (int i = 0; i < 4; ++i)
#pragma unroll
      for (int j = 0; j < 2; ++j) {
        acc[i][j] = wmma_bf16(fAl[i], fBh[j], acc[i][j]);  // small terms first
        acc[i][j] = wmma_bf16(fAh[i], fBl[j], acc[i][j]);
        acc[i][j] = wmma_bf16(fAh[i], fBh[j], acc[i][j]);
      }
    __syncthreads();                       // protect bf16 tiles for next pass
    buf ^= 1;
  }

  // epilogue: C/D layout -> lane holds col n=l&15, rows v + 8*(l>>4)
#pragma unroll
  for (int i = 0; i < 4; ++i)
#pragma unroll
    for (int j = 0; j < 2; ++j) {
      int n = tn * TN + wn * 32 + j * 16 + lr;
      float bv = bias[e * DFF + n];
      float* op = out + ((size_t)e * CAP + tm * TM + wm * 64 + i * 16) * DFF + n;
#pragma unroll
      for (int v = 0; v < 8; ++v) {
        int m_off = v + half * 8;
        op[(size_t)m_off * DFF] = acc[i][j][v] + bv;
      }
    }
}

// ---------------------------------------------------------------------------
extern "C" void kernel_launch(void* const* d_in, const int* in_sizes, int n_in,
                              void* d_out, int out_size, void* d_ws, size_t ws_size,
                              hipStream_t stream) {
  const float* x     = (const float*)d_in[0];
  const float* gw    = (const float*)d_in[1];
  const float* ew    = (const float*)d_in[2];
  const float* ebias = (const float*)d_in[3];

  float* out  = (float*)d_out;                    // [E][CAP][DFF]
  float* cw   = out + OUT_ELEMS;                  // [TOKENS][E][CAP]
  float* laux = cw + CW_ELEMS;                    // scalar

  int*   top1   = (int*)d_ws;                     // TOKENS
  float* gate_w = (float*)(top1 + TOKENS);        // TOKENS
  float* wsme   = gate_w + TOKENS;                // (TOKENS/8)*NEXP
  int*   slots  = (int*)(wsme + TOKENS);          // NSLOTS
  int*   dest   = slots + TOKENS;                 // placed after slots
  // note: slots region is NSLOTS ints; dest must start after it
  dest = slots + NSLOTS;

  init_kernel<<<4096, 256, 0, stream>>>(cw, CW_ELEMS / 4, slots, NSLOTS);
  gate_kernel<<<TOKENS / 8, 256, 0, stream>>>(x, gw, top1, gate_w, wsme);
  rank_kernel<<<1, 256, 0, stream>>>(top1, wsme, slots, dest, laux);
  scatter_cw<<<TOKENS / 256, 256, 0, stream>>>(dest, gate_w, cw);
  moe_gemm<<<NEXP * (CAP / TM) * (DFF / TN), 256, 0, stream>>>(
      x, ew, ebias, slots, out);
}
